// MultiHeadAttention_798863917665
// MI455X (gfx1250) — matrix operand order repaired and checked by reading, not compiled, not run on hardware
//
#include <hip/hip_runtime.h>

// ---------------------------------------------------------------------------
// MI455X / gfx1250 multi-head attention, bf16 WMMA (f32 accumulate), wave32.
//   1) fp32 -> bf16 conversion of activations + weights (workspace)
//   2) Q/K/V projections: bf16 GEMM (A[M,K] x W[N,K]^T + bias), wave tile
//      32x64 (2x4 WMMA tiles), epilogue scatters to split-head layouts
//      (V stored depth-major per head so PV fragments are contiguous)
//   3) flash-attention: one wave per 16-query tile, 64 keys per iteration
//      (4 S-tiles -> one online-softmax pass), P re-laid out via per-wave LDS
//   4) output projection GEMM -> fp32 d_out
// ---------------------------------------------------------------------------

typedef __bf16 bf16;
typedef __attribute__((ext_vector_type(16))) __bf16 v16bf;
typedef __attribute__((ext_vector_type(8)))  float  v8f;

union Frag16 {
    v16bf v;
    uint4 u[2];
};

__device__ __forceinline__ v8f wmma_bf16(v16bf a, v16bf b, v8f c) {
    // D = A(16x32) * B(32x16) + C, f32 accumulate
    return __builtin_amdgcn_wmma_f32_16x16x32_bf16(
        /*neg_a=*/false, a, /*neg_b=*/false, b,
        /*c_mod=*/(short)0, c, /*reuse_a=*/false, /*reuse_b=*/false);
}

// ---------------------------------------------------------------------------
// fp32 -> bf16 elementwise conversion
// ---------------------------------------------------------------------------
__global__ void mha_cvt_f32_bf16(const float* __restrict__ src,
                                 bf16* __restrict__ dst, int n) {
    int i = blockIdx.x * blockDim.x + threadIdx.x;
    int stride = gridDim.x * blockDim.x;
    for (; i < n; i += stride) dst[i] = (bf16)src[i];
}

// ---------------------------------------------------------------------------
// GEMM: Y[M,N] = A[M,K] @ W[N,K]^T + bias[N]
//   layout 0: outF fp32 row-major [M,N]
//   layout 1: outB bf16 split-heads   [b,h,s,d]
//   layout 2: outB bf16 split-heads-T [b,h,d,s]
// Wave tile 32x64 (2x4 WMMA tiles, 8 WMMAs / 32-K step); 8 waves -> 64x256.
// ---------------------------------------------------------------------------
__global__ __launch_bounds__(256)
void mha_gemm_bf16(const bf16* __restrict__ A, const bf16* __restrict__ W,
                   const float* __restrict__ bias,
                   bf16* __restrict__ outB, float* __restrict__ outF,
                   int M, int N, int K, int layout) {
    const int lane = threadIdx.x & 31;
    const int wid  = threadIdx.x >> 5;
    const int wm = wid >> 2, wn = wid & 3;
    const int m0 = blockIdx.y * 64 + wm * 32;
    const int n0 = blockIdx.x * 256 + wn * 64;

    const int c16  = lane & 15;
    const int dofs = (lane & 16) ? 16 : 0;  // A-frag K sub-offset
    const int kb   = (lane & 16) ? 16 : 0;  // B-frag K base

    const bf16* a0p = A + (size_t)(m0 + c16) * K + dofs;
    const bf16* a1p = a0p + (size_t)16 * K;
    const bf16* bp0 = W + (size_t)(n0 + c16) * K + kb;
    const bf16* bp1 = bp0 + (size_t)16 * K;
    const bf16* bp2 = bp0 + (size_t)32 * K;
    const bf16* bp3 = bp0 + (size_t)48 * K;

    v8f acc[2][4] = {};
    for (int k0 = 0; k0 < K; k0 += 32) {
        Frag16 a0, a1, b[4];
        a0.u[0] = *(const uint4*)(a0p + k0);
        a0.u[1] = *(const uint4*)(a0p + k0 + 8);
        a1.u[0] = *(const uint4*)(a1p + k0);
        a1.u[1] = *(const uint4*)(a1p + k0 + 8);
        b[0].u[0] = *(const uint4*)(bp0 + k0);
        b[0].u[1] = *(const uint4*)(bp0 + k0 + 8);
        b[1].u[0] = *(const uint4*)(bp1 + k0);
        b[1].u[1] = *(const uint4*)(bp1 + k0 + 8);
        b[2].u[0] = *(const uint4*)(bp2 + k0);
        b[2].u[1] = *(const uint4*)(bp2 + k0 + 8);
        b[3].u[0] = *(const uint4*)(bp3 + k0);
        b[3].u[1] = *(const uint4*)(bp3 + k0 + 8);
        __builtin_prefetch(a0p + k0 + 128, 0, 3);   // WGP-scope prefetch
        __builtin_prefetch(bp0 + k0 + 128, 0, 3);
        __builtin_prefetch(bp2 + k0 + 128, 0, 3);
#pragma unroll
        for (int j = 0; j < 4; ++j) {
            acc[0][j] = wmma_bf16(a0.v, b[j].v, acc[0][j]);
            acc[1][j] = wmma_bf16(a1.v, b[j].v, acc[1][j]);
        }
    }

    const int rb = (lane & 16) ? 8 : 0;
#pragma unroll
    for (int i = 0; i < 2; ++i) {
#pragma unroll
        for (int j = 0; j < 4; ++j) {
#pragma unroll
            for (int r = 0; r < 8; ++r) {
                const int row = m0 + 16 * i + rb + r;
                const int col = n0 + 16 * j + c16;
                const float val = acc[i][j][r] + bias[col];
                if (layout == 0) {
                    outF[(size_t)row * N + col] = val;
                } else {
                    const int bb = row >> 11, s = row & 2047;
                    const int h = col >> 6, d = col & 63;
                    if (layout == 1)
                        outB[(((size_t)bb * 16 + h) * 2048 + s) * 64 + d] = (bf16)val;
                    else
                        outB[(((size_t)bb * 16 + h) * 64 + d) * 2048 + s] = (bf16)val;
                }
            }
        }
    }
}

// ---------------------------------------------------------------------------
// Flash attention: one wave handles one 16-query tile of one (b,h).
// Qh,Kh: [b,h,s,64] bf16.  Vt: [b,h,64,s] bf16 (depth-major).
// 64 keys per iteration: 4 S-tiles share one online-softmax pass.
// Obf out: [b*s, h*64+d] bf16 row-major [4096,1024].
// ---------------------------------------------------------------------------
__global__ __launch_bounds__(256)
void mha_attn_fwd(const bf16* __restrict__ Qh, const bf16* __restrict__ Kh,
                  const bf16* __restrict__ Vt, bf16* __restrict__ Obf) {
    __shared__ __align__(16) bf16 plds[8][16 * 64];   // per-wave P tile, 16 KB

    const int lane = threadIdx.x & 31;
    const int wid  = threadIdx.x >> 5;
    const int tile = blockIdx.x * 8 + wid;   // 0..4095
    const int bh = tile >> 7;                // (b*16+h), 0..31
    const int q0 = (tile & 127) * 16;

    const bf16* Qb = Qh + (size_t)bh * 2048 * 64;
    const bf16* Kb = Kh + (size_t)bh * 2048 * 64;
    const bf16* Vb = Vt + (size_t)bh * 64 * 2048;

    const int c16   = lane & 15;
    const int dofs  = (lane & 16) ? 16 : 0;
    const int kb16  = (lane & 16) ? 16 : 0;
    const int rbase = (lane & 16) ? 8 : 0;

    // Q A-fragments (loop-invariant): depth chunks 0..31 and 32..63
    Frag16 aq0, aq1;
    {
        const bf16* qp = Qb + (size_t)(q0 + c16) * 64 + dofs;
        aq0.u[0] = *(const uint4*)(qp);
        aq0.u[1] = *(const uint4*)(qp + 8);
        aq1.u[0] = *(const uint4*)(qp + 32);
        aq1.u[1] = *(const uint4*)(qp + 40);
    }

    float m[8], l[8];
    v8f o[4] = {};
#pragma unroll
    for (int r = 0; r < 8; ++r) { m[r] = -__builtin_inff(); l[r] = 0.0f; }

    const float scale = 0.125f;  // 1/sqrt(64)
    bf16* myp = &plds[wid][0];

    for (int kc = 0; kc < 2048; kc += 64) {
        // ---- S = Q @ K^T : four 16-key tiles, depth 64 = 2 WMMAs each
        v8f s[4];
#pragma unroll
        for (int st = 0; st < 4; ++st) {
            const bf16* kp = Kb + (size_t)(kc + 16 * st + c16) * 64 + kb16;
            Frag16 bk;
            v8f acc = {};
            bk.u[0] = *(const uint4*)(kp);      bk.u[1] = *(const uint4*)(kp + 8);
            acc = wmma_bf16(aq0.v, bk.v, acc);
            bk.u[0] = *(const uint4*)(kp + 32); bk.u[1] = *(const uint4*)(kp + 40);
            acc = wmma_bf16(aq1.v, bk.v, acc);
            s[st] = acc;
        }
        __builtin_prefetch(Kb + (size_t)(kc + 64 + c16) * 64, 0, 3);
        __builtin_prefetch(Vb + (size_t)c16 * 2048 + kc + 64, 0, 3);

        // ---- one online-softmax pass for all 64 keys
#pragma unroll
        for (int r = 0; r < 8; ++r) {
            const float x0 = s[0][r] * scale, x1 = s[1][r] * scale;
            const float x2 = s[2][r] * scale, x3 = s[3][r] * scale;
            float t = fmaxf(fmaxf(x0, x1), fmaxf(x2, x3));
            t = fmaxf(t, __shfl_xor(t, 1, 32));
            t = fmaxf(t, __shfl_xor(t, 2, 32));
            t = fmaxf(t, __shfl_xor(t, 4, 32));
            t = fmaxf(t, __shfl_xor(t, 8, 32));
            const float mnew = fmaxf(m[r], t);
            const float corr = __expf(m[r] - mnew);
            const float p0 = __expf(x0 - mnew);
            const float p1 = __expf(x1 - mnew);
            const float p2 = __expf(x2 - mnew);
            const float p3 = __expf(x3 - mnew);
            float rs = (p0 + p1) + (p2 + p3);
            rs += __shfl_xor(rs, 1, 32);
            rs += __shfl_xor(rs, 2, 32);
            rs += __shfl_xor(rs, 4, 32);
            rs += __shfl_xor(rs, 8, 32);
            l[r] = l[r] * corr + rs;
            m[r] = mnew;
            o[0][r] = o[0][r] * corr;
            o[1][r] = o[1][r] * corr;
            o[2][r] = o[2][r] * corr;
            o[3][r] = o[3][r] * corr;
            // relayout P (D-frag) -> row-major 16x64 in LDS (same-wave, DScnt-ordered)
            bf16* prow = myp + (size_t)(rbase + r) * 64 + c16;
            prow[0]  = (bf16)p0;
            prow[16] = (bf16)p1;
            prow[32] = (bf16)p2;
            prow[48] = (bf16)p3;
        }

        // ---- reload P as two A-fragments (16 queries x 32 keys each)
        Frag16 pa0, pa1;
        {
            const bf16* pr = myp + (size_t)c16 * 64 + dofs;
            pa0.u[0] = *(const uint4*)(pr);
            pa0.u[1] = *(const uint4*)(pr + 8);
            pa1.u[0] = *(const uint4*)(pr + 32);
            pa1.u[1] = *(const uint4*)(pr + 40);
        }

        // ---- O += P @ V : B-frags contiguous thanks to depth-major V
#pragma unroll
        for (int j = 0; j < 4; ++j) {
            const bf16* vpj = Vb + (size_t)(16 * j + c16) * 2048 + kc + kb16;
            Frag16 fv;
            fv.u[0] = *(const uint4*)(vpj);      fv.u[1] = *(const uint4*)(vpj + 8);
            o[j] = wmma_bf16(pa0.v, fv.v, o[j]);
            fv.u[0] = *(const uint4*)(vpj + 32); fv.u[1] = *(const uint4*)(vpj + 40);
            o[j] = wmma_bf16(pa1.v, fv.v, o[j]);
        }
    }

    // ---- normalize and store to [b*s, h*64+d] bf16 for the output projection
    const int b = bh >> 4, h = bh & 15;
#pragma unroll
    for (int r = 0; r < 8; ++r) {
        const float inv = 1.0f / l[r];
        const size_t orow = (size_t)(b * 2048 + q0 + rbase + r) * 1024 + h * 64 + c16;
        Obf[orow +  0] = (bf16)(o[0][r] * inv);
        Obf[orow + 16] = (bf16)(o[1][r] * inv);
        Obf[orow + 32] = (bf16)(o[2][r] * inv);
        Obf[orow + 48] = (bf16)(o[3][r] * inv);
    }
}

// ---------------------------------------------------------------------------
extern "C" void kernel_launch(void* const* d_in, const int* in_sizes, int n_in,
                              void* d_out, int out_size, void* d_ws, size_t ws_size,
                              hipStream_t stream) {
    const float* q  = (const float*)d_in[0];
    const float* k  = (const float*)d_in[1];
    const float* v  = (const float*)d_in[2];
    const float* wq = (const float*)d_in[3];
    const float* bq = (const float*)d_in[4];
    const float* wk = (const float*)d_in[5];
    const float* bk = (const float*)d_in[6];
    const float* wv = (const float*)d_in[7];
    const float* bv = (const float*)d_in[8];
    const float* wo = (const float*)d_in[9];
    const float* bo = (const float*)d_in[10];
    float* out = (float*)d_out;

    const int MSZ = 4096 * 1024;   // activation elements (B*S x D_MODEL)
    const int WSZ = 1024 * 1024;   // weight elements

    bf16* p = (bf16*)d_ws;         // total 64 MB of bf16 scratch
    bf16* qbf = p;  p += MSZ;
    bf16* kbf = p;  p += MSZ;
    bf16* vbf = p;  p += MSZ;
    bf16* wqb = p;  p += WSZ;
    bf16* wkb = p;  p += WSZ;
    bf16* wvb = p;  p += WSZ;
    bf16* wob = p;  p += WSZ;
    bf16* Qh  = p;  p += MSZ;      // [b,h,s,d]
    bf16* Kh  = p;  p += MSZ;      // [b,h,s,d]
    bf16* Vt  = p;  p += MSZ;      // [b,h,d,s]
    bf16* Ob  = p;  p += MSZ;      // [b*s, h*64+d]

    mha_cvt_f32_bf16<<<1024, 256, 0, stream>>>(q,  qbf, MSZ);
    mha_cvt_f32_bf16<<<1024, 256, 0, stream>>>(k,  kbf, MSZ);
    mha_cvt_f32_bf16<<<1024, 256, 0, stream>>>(v,  vbf, MSZ);
    mha_cvt_f32_bf16<<<512,  256, 0, stream>>>(wq, wqb, WSZ);
    mha_cvt_f32_bf16<<<512,  256, 0, stream>>>(wk, wkb, WSZ);
    mha_cvt_f32_bf16<<<512,  256, 0, stream>>>(wv, wvb, WSZ);
    mha_cvt_f32_bf16<<<512,  256, 0, stream>>>(wo, wob, WSZ);

    dim3 gg(1024 / 256, 4096 / 64);  // (4, 64)
    mha_gemm_bf16<<<gg, 256, 0, stream>>>(qbf, wqb, bq, Qh, nullptr, 4096, 1024, 1024, 1);
    mha_gemm_bf16<<<gg, 256, 0, stream>>>(kbf, wkb, bk, Kh, nullptr, 4096, 1024, 1024, 1);
    mha_gemm_bf16<<<gg, 256, 0, stream>>>(vbf, wvb, bv, Vt, nullptr, 4096, 1024, 1024, 2);

    mha_attn_fwd<<<512, 256, 0, stream>>>(Qh, Kh, Vt, Ob);

    mha_gemm_bf16<<<gg, 256, 0, stream>>>(Ob, wob, bo, nullptr, out, 4096, 1024, 1024, 0);
}
